// GCN_75685913690132
// MI455X (gfx1250) — matrix operand order repaired and checked by reading, not compiled, not run on hardware
//
#include <hip/hip_runtime.h>
#include <hip/hip_bf16.h>

// ---------------------------------------------------------------------------
// 2-layer GCN forward for gfx1250 (MI455X).
//   layer1: h1 = x@W1 (bf16 WMMA) ; agg over edges1 with sym-norm ; +b1 = hidden
//   BN(batch stats) + ReLU -> bf16
//   layer2: h2 = hrelu@W2 (bf16 WMMA) ; agg over edges2 ; +b2 = out
// d_out = [out (N*64) | hidden (N*128)] fp32.
// ---------------------------------------------------------------------------

typedef __attribute__((ext_vector_type(16))) __bf16 v16bf;
typedef __attribute__((ext_vector_type(8)))  float  v8f;

static constexpr int KDIM = 128;   // IN == HID == 128 (K for both GEMMs)
static constexpr int KCH  = KDIM / 32;

// ---- degree / normalization -----------------------------------------------

__global__ void gcn_deg_init(float* deg1, float* deg2, float* stats, int n) {
    int i = blockIdx.x * blockDim.x + threadIdx.x;
    if (i < n) { deg1[i] = 1.0f; deg2[i] = 1.0f; }   // self-loop
    if (i < 256) stats[i] = 0.0f;                    // gsum/gsumsq
}

__global__ void gcn_deg_acc(const int* ei, float* deg, int e) {
    int i = blockIdx.x * blockDim.x + threadIdx.x;
    if (i < e) atomicAdd(&deg[ei[e + i]], 1.0f);     // dst row of edge_index
}

__global__ void gcn_dinv(float* deg, int n) {
    int i = blockIdx.x * blockDim.x + threadIdx.x;
    if (i < n) deg[i] = rsqrtf(deg[i]);              // deg >= 1 always
}

// ---- bf16 packing into WMMA lane layouts ----------------------------------
// A 16x32 bf16 layout (per ISA 7.12.2): lane<16 -> K {0..7,16..23}, lane>=16 -> K {8..15,24..31}
__device__ __forceinline__ int a_kmap(int lane, int j) {
    int base = (lane < 16) ? 0 : 8;
    return (j < 8) ? (base + j) : (16 + base + (j - 8));
}

// Xp[((strip*KCH + kc)*32 + lane)*16 + j] = bf16(X[row][k])
__global__ void gcn_pack_a(const float* X, __bf16* Xp, int total) {
    int idx = blockIdx.x * blockDim.x + threadIdx.x;
    if (idx >= total) return;
    int j = idx & 15, lane = (idx >> 4) & 31, t = idx >> 9;
    int kc = t & (KCH - 1), strip = t >> 2;          // KCH == 4
    int row = strip * 16 + (lane & 15);
    int k   = kc * 32 + a_kmap(lane, j);
    Xp[idx] = (__bf16)X[(size_t)row * KDIM + k];
}

// B 32x16: same K order as A (col = lane&15): lane<16 -> K {0..7,16..23}, lane>=16 -> K {8..15,24..31}
__global__ void gcn_pack_w(const float* W, __bf16* Wp, int nc, int total) {
    int idx = blockIdx.x * blockDim.x + threadIdx.x;
    if (idx >= total) return;
    int j = idx & 15, lane = (idx >> 4) & 31, t = idx >> 9;
    int kc = t & (KCH - 1), ct = t >> 2;
    int col = ct * 16 + (lane & 15);
    int k   = kc * 32 + a_kmap(lane, j);
    Wp[idx] = (__bf16)W[(size_t)k * nc + col];
}

// ---- WMMA GEMM: [N,128](bf16,A-packed) x [128,NC](bf16,B-packed) -> [N,NC] f32
template <int NCT>
__global__ __launch_bounds__(256)
void gcn_gemm_wmma(const __bf16* __restrict__ Ap, const __bf16* __restrict__ Wp,
                   float* __restrict__ C, int strips, int nc) {
    int wave = (blockIdx.x * blockDim.x + threadIdx.x) >> 5;
    int lane = threadIdx.x & 31;
    if (wave >= strips) return;

    const v16bf* Abase = reinterpret_cast<const v16bf*>(Ap) + (size_t)wave * KCH * 32;
    v16bf a[KCH];
#pragma unroll
    for (int kc = 0; kc < KCH; ++kc) a[kc] = Abase[kc * 32 + lane];

    int col0  = lane & 15;
    int rbase = wave * 16 + ((lane < 16) ? 0 : 8);
#pragma unroll
    for (int ct = 0; ct < NCT; ++ct) {
        const v16bf* Bbase = reinterpret_cast<const v16bf*>(Wp) + (size_t)ct * KCH * 32;
        v8f acc = {};
#pragma unroll
        for (int kc = 0; kc < KCH; ++kc) {
            v16bf b = Bbase[kc * 32 + lane];
            acc = __builtin_amdgcn_wmma_f32_16x16x32_bf16(
                false, a[kc], false, b, (short)0, acc, false, false);
        }
        int col = ct * 16 + col0;
#pragma unroll
        for (int r = 0; r < 8; ++r)
            C[(size_t)(rbase + r) * nc + col] = acc[r];
    }
}

// ---- self-loop seed: out = h*dinv^2 + bias --------------------------------
template <int LOGC>
__global__ void gcn_seed(const float* __restrict__ h, const float* __restrict__ dinv,
                         const float* __restrict__ bias, float* __restrict__ out, int total) {
    int idx = blockIdx.x * blockDim.x + threadIdx.x;
    if (idx >= total) return;
    int i = idx >> LOGC, c = idx & ((1 << LOGC) - 1);
    float di = dinv[i];
    out[idx] = h[idx] * di * di + bias[c];
}

// ---- edge scatter: out[dst] += h[src] * dinv[src]*dinv[dst] ---------------
template <int C>
__global__ void gcn_scatter(const int* __restrict__ ei, const float* __restrict__ h,
                            const float* __restrict__ dinv, float* __restrict__ out, int e) {
    constexpr int PER = C / 4;                  // threads per edge (float4 each)
    constexpr int SH  = (C == 128) ? 5 : 4;
    int idx = blockIdx.x * blockDim.x + threadIdx.x;
    int eidx = idx >> SH;
    if (eidx >= e) return;
    int c4 = (idx & (PER - 1)) * 4;
    int s = ei[eidx], d = ei[e + eidx];
    float w = dinv[s] * dinv[d];
    const float4 v = *reinterpret_cast<const float4*>(h + (size_t)s * C + c4);
    float* o = out + (size_t)d * C + c4;
    atomicAdd(o + 0, v.x * w);
    atomicAdd(o + 1, v.y * w);
    atomicAdd(o + 2, v.z * w);
    atomicAdd(o + 3, v.w * w);
}

// ---- batch-norm statistics ------------------------------------------------
__global__ void gcn_bn_stats(const float* __restrict__ hidden, float* gsum, float* gsq, int n) {
    int c = threadIdx.x & 127;
    int half = threadIdx.x >> 7;                 // blockDim = 256 -> 2 rows in flight
    float s = 0.f, q = 0.f;
    for (int i = blockIdx.x * 2 + half; i < n; i += gridDim.x * 2) {
        float v = hidden[(size_t)i * 128 + c];
        s += v; q += v * v;
    }
    __shared__ float ss[256], sq[256];
    ss[threadIdx.x] = s; sq[threadIdx.x] = q;
    __syncthreads();
    if (threadIdx.x < 128) {
        atomicAdd(&gsum[c], ss[threadIdx.x] + ss[threadIdx.x + 128]);
        atomicAdd(&gsq[c],  sq[threadIdx.x] + sq[threadIdx.x + 128]);
    }
}

__global__ void gcn_bn_finalize(const float* gsum, const float* gsq,
                                const float* gamma, const float* beta,
                                float* scale, float* shift, float invN) {
    int c = threadIdx.x;                         // 128 threads
    float mean = gsum[c] * invN;
    float var  = gsq[c] * invN - mean * mean;
    float sc   = gamma[c] * rsqrtf(var + 1e-5f);
    scale[c] = sc;
    shift[c] = beta[c] - mean * sc;
}

// ---- BN + ReLU + repack into A layout (bf16) ------------------------------
__global__ void gcn_bn_relu_pack(const float* __restrict__ hidden,
                                 const float* __restrict__ scale,
                                 const float* __restrict__ shift,
                                 __bf16* __restrict__ Xp, int total) {
    int idx = blockIdx.x * blockDim.x + threadIdx.x;
    if (idx >= total) return;
    int j = idx & 15, lane = (idx >> 4) & 31, t = idx >> 9;
    int kc = t & (KCH - 1), strip = t >> 2;
    int row = strip * 16 + (lane & 15);
    int k   = kc * 32 + a_kmap(lane, j);
    float v = hidden[(size_t)row * 128 + k];
    v = v * scale[k] + shift[k];
    v = v > 0.f ? v : 0.f;
    Xp[idx] = (__bf16)v;
}

// ---------------------------------------------------------------------------

extern "C" void kernel_launch(void* const* d_in, const int* in_sizes, int n_in,
                              void* d_out, int out_size, void* d_ws, size_t ws_size,
                              hipStream_t stream) {
    const float* x     = (const float*)d_in[0];
    const int*   ei1   = (const int*)d_in[1];
    const int*   ei2   = (const int*)d_in[2];
    const float* W1    = (const float*)d_in[3];
    const float* b1    = (const float*)d_in[4];
    const float* gamma = (const float*)d_in[5];
    const float* beta  = (const float*)d_in[6];
    const float* W2    = (const float*)d_in[7];
    const float* b2    = (const float*)d_in[8];

    const int N  = in_sizes[0] / 128;    // 50000
    const int E1 = in_sizes[1] / 2;      // 800000
    const int E2 = in_sizes[2] / 2;

    float* out_p    = (float*)d_out;                 // [N,64]
    float* hidden_p = out_p + (size_t)N * 64;        // [N,128]

    // workspace carve (256B aligned slices)
    char*  base = (char*)d_ws;
    size_t off  = 0;
    auto carve = [&](size_t bytes) -> void* {
        void* p = base + off;
        off = (off + bytes + 255) & ~(size_t)255;
        return p;
    };
    float*  deg1  = (float*)carve((size_t)N * 4);
    float*  deg2  = (float*)carve((size_t)N * 4);
    float*  stats = (float*)carve(256 * 4);          // gsum[128] | gsq[128]
    float*  scale = (float*)carve(128 * 4);
    float*  shift = (float*)carve(128 * 4);
    __bf16* xp    = (__bf16*)carve((size_t)N * 128 * 2);
    __bf16* w1p   = (__bf16*)carve((size_t)128 * 128 * 2);
    __bf16* w2p   = (__bf16*)carve((size_t)128 * 64 * 2);
    float*  h1    = (float*)carve((size_t)N * 128 * 4);
    __bf16* hrp   = (__bf16*)carve((size_t)N * 128 * 2);
    float*  h2    = (float*)carve((size_t)N * 64 * 4);
    float*  gsum  = stats;
    float*  gsq   = stats + 128;

    const int B = 256;
    auto nb = [](long long total, int b) { return (int)((total + b - 1) / b); };

    // degrees + normalization
    gcn_deg_init<<<nb(N, B), B, 0, stream>>>(deg1, deg2, stats, N);
    gcn_deg_acc<<<nb(E1, B), B, 0, stream>>>(ei1, deg1, E1);
    gcn_deg_acc<<<nb(E2, B), B, 0, stream>>>(ei2, deg2, E2);
    gcn_dinv<<<nb(N, B), B, 0, stream>>>(deg1, N);
    gcn_dinv<<<nb(N, B), B, 0, stream>>>(deg2, N);

    // pack operands for WMMA
    gcn_pack_a<<<nb((long long)N * 128, B), B, 0, stream>>>(x, xp, N * 128);
    gcn_pack_w<<<nb(128 * 128, B), B, 0, stream>>>(W1, w1p, 128, 128 * 128);
    gcn_pack_w<<<nb(128 * 64, B), B, 0, stream>>>(W2, w2p, 64, 128 * 64);

    // layer 1: GEMM -> seed(self-loop + bias) -> edge scatter
    const int strips = N / 16;                       // 3125
    gcn_gemm_wmma<8><<<nb((long long)strips * 32, B), B, 0, stream>>>(xp, w1p, h1, strips, 128);
    gcn_seed<7><<<nb((long long)N * 128, B), B, 0, stream>>>(h1, deg1, b1, hidden_p, N * 128);
    gcn_scatter<128><<<nb((long long)E1 * 32, B), B, 0, stream>>>(ei1, h1, deg1, hidden_p, E1);

    // batch norm (batch stats) + ReLU, repacked to bf16 A layout
    gcn_bn_stats<<<512, 256, 0, stream>>>(hidden_p, gsum, gsq, N);
    gcn_bn_finalize<<<1, 128, 0, stream>>>(gsum, gsq, gamma, beta, scale, shift, 1.0f / (float)N);
    gcn_bn_relu_pack<<<nb((long long)N * 128, B), B, 0, stream>>>(hidden_p, scale, shift, hrp, N * 128);

    // layer 2: GEMM -> seed -> edge scatter
    gcn_gemm_wmma<4><<<nb((long long)strips * 32, B), B, 0, stream>>>(hrp, w2p, h2, strips, 64);
    gcn_seed<6><<<nb((long long)N * 64, B), B, 0, stream>>>(h2, deg2, b2, out_p, N * 64);
    gcn_scatter<64><<<nb((long long)E2 * 16, B), B, 0, stream>>>(ei2, h2, deg2, out_p, E2);
}
